// AxialTimeAttention_17789754540075
// MI455X (gfx1250) — compile-verified, matrix operand order checked by reading, not run on hardware
//
#include <hip/hip_runtime.h>
#include <hip/hip_bf16.h>
#include <math.h>

// ---------------------------------------------------------------------------
// Problem constants (from reference)
// ---------------------------------------------------------------------------
#define TT 64          // time steps
#define CC 768         // channels
#define SS 512         // H*W*D spatial
#define HE 12          // heads
#define HD 64          // head dim
#define N_QKV (3*CC)   // 2304
#define MM (TT*SS)     // 32768 rows of the activation GEMMs

typedef __attribute__((ext_vector_type(16))) __bf16 bf16x16;
typedef __attribute__((ext_vector_type(8)))  float  f32x8;

// ---------------------------------------------------------------------------
// WMMA helpers (CDNA5 wave32, v_wmma_f32_16x16x32_bf16)
// ---------------------------------------------------------------------------
__device__ __forceinline__ f32x8 wmma_bf16(bf16x16 a, bf16x16 b, f32x8 c) {
  return __builtin_amdgcn_wmma_f32_16x16x32_bf16(
      /*neg_a=*/false, a, /*neg_b=*/false, b,
      /*c_mod=*/(short)0, c, /*reuse_a=*/false, /*reuse_b=*/false);
}

// Load a 16(row) x 32(K) bf16 fragment from LDS (row-major, leading dim ld).
// ISA 16-bit A layout: lane half h=lane>>4, row=lane&15,
// VGPR v holds K = 16*(v>=4) + 2*(v&3) + 8*h and K+1.
// Works for both A (rows=M) and B (rows=N; B stored transposed = NT GEMM).
__device__ __forceinline__ bf16x16 load_frag16(const __bf16* base, int ld, int lane) {
  const int r = lane & 15;
  const int h = (lane >> 4) & 1;
  bf16x16 f;
#pragma unroll
  for (int v = 0; v < 8; ++v) {
    const int kb = ((v >> 2) << 4) + ((v & 3) << 1) + (h << 3);
    f[2 * v]     = base[r * ld + kb];
    f[2 * v + 1] = base[r * ld + kb + 1];
  }
  return f;
}

// ---------------------------------------------------------------------------
// Prep: fp32 -> bf16 cast (weights), rel-pos bias table
// ---------------------------------------------------------------------------
__global__ void cast_bf16_kernel(const float* __restrict__ src,
                                 __bf16* __restrict__ dst, int n) {
  int i = blockIdx.x * 256 + threadIdx.x;
  if (i < n) dst[i] = (__bf16)src[i];
}

// biasT[he][i][j] = rel_emb[bucket(i,j)][he], T5 bidirectional, 32 buckets,
// max_distance=128.  relpos[i][j] = j - i, n = -relpos = i - j.
__global__ void relbias_kernel(const float* __restrict__ rel_emb,
                               float* __restrict__ biasT) {
  int i = blockIdx.x;    // 0..63 context
  int j = threadIdx.x;   // 0..63 memory
  int n = i - j;
  int ret = (n < 0) ? 16 : 0;
  int na = (n < 0) ? -n : n;
  int bucket;
  if (na < 8) {
    bucket = ret + na;
  } else {
    int vl = 8 + (int)(logf((float)na * 0.125f) * (8.0f / logf(16.0f)));
    if (vl > 15) vl = 15;
    bucket = ret + vl;
  }
#pragma unroll
  for (int he = 0; he < HE; ++he)
    biasT[(size_t)he * (TT * TT) + i * TT + j] = rel_emb[bucket * HE + he];
}

// ---------------------------------------------------------------------------
// RMSGroupNorm: per (tb,c) RMS over 512 spatial; emit bf16 activation matrix
// xnA[m=tb*512+s][k=c]  (row-major K=768) for the QKV GEMM.
// ---------------------------------------------------------------------------
__global__ __launch_bounds__(128)
void rmsnorm_kernel(const float* __restrict__ x, const float* __restrict__ nw,
                    __bf16* __restrict__ xnA) {
  const int c = blockIdx.x;   // 0..767
  const int tb = blockIdx.y;  // 0..63
  const float* row = x + ((size_t)tb * CC + c) * SS;
  float ss = 0.f;
  for (int i = threadIdx.x; i < SS; i += 128) {
    float v = row[i];
    ss += v * v;
  }
  __shared__ float red[128];
  red[threadIdx.x] = ss;
  __syncthreads();
  for (int o = 64; o > 0; o >>= 1) {
    if (threadIdx.x < o) red[threadIdx.x] += red[threadIdx.x + o];
    __syncthreads();
  }
  const float scale = nw[c] * rsqrtf(red[0] * (1.f / SS) + 1e-6f);
  for (int i = threadIdx.x; i < SS; i += 128)
    xnA[((size_t)tb * SS + i) * CC + c] = (__bf16)(row[i] * scale);
}

// ---------------------------------------------------------------------------
// NT GEMM: C[m][n] = sum_k A[m][k]*Bt[n][k], K = 768.
// 128x128 block tile, 256 threads = 8 waves in a 4x2 grid; each wave computes
// a 32x64 sub-tile (2 A-frags x 4 B-frags = 8 WMMAs per 32-K step).
// LDS double-buffered: one barrier per K-step.
// MODE 0: fp32 out, +bias[n], row-major ldc=Nld           (QKV projection)
// MODE 1: +bias[n]+residual, scatter to (T,C,S) layout    (output projection)
// ---------------------------------------------------------------------------
template <int MODE>
__global__ __launch_bounds__(256)
void gemm_nt_kernel(const __bf16* __restrict__ A, const __bf16* __restrict__ Bt,
                    const float* __restrict__ bias, const float* __restrict__ resid,
                    float* __restrict__ Cout, int Nld) {
  const int m0 = blockIdx.x * 128;
  const int n0 = blockIdx.y * 128;
  __shared__ __bf16 As[2][128][40];  // +8 pad, 10 KB per buffer
  __shared__ __bf16 Bs[2][128][40];
  const int tid  = threadIdx.x;
  const int lane = tid & 31;
  const int wv   = tid >> 5;
  const int wr   = wv & 3;   // wave m-strip (32 rows each)
  const int wc   = wv >> 2;  // wave n-half  (64 cols each)

  // staging registers: 128x32 bf16 tile = 1024 quads of 4 bf16, 4 per thread
  uint2 ar[4], br[4];
  auto gload = [&](int kbase) {
#pragma unroll
    for (int i = 0; i < 4; ++i) {
      const int q = tid + (i << 8);
      const int r = q >> 3;
      const int kq = (q & 7) << 2;
      ar[i] = *(const uint2*)&A [(size_t)(m0 + r) * CC + kbase + kq];
      br[i] = *(const uint2*)&Bt[(size_t)(n0 + r) * CC + kbase + kq];
    }
  };
  auto lstore = [&](int pp) {
#pragma unroll
    for (int i = 0; i < 4; ++i) {
      const int q = tid + (i << 8);
      const int r = q >> 3;
      const int kq = (q & 7) << 2;
      *(uint2*)&As[pp][r][kq] = ar[i];
      *(uint2*)&Bs[pp][r][kq] = br[i];
    }
  };

  f32x8 acc[8] = {};

  gload(0);
  lstore(0);
  __syncthreads();

  int p = 0;
  for (int k0 = 0; k0 < CC; k0 += 32) {
    // next tile (clamped wrap on last iter: redundant but branch-free)
    const int knext = (k0 + 32 < CC) ? (k0 + 32) : 0;
    gload(knext);
    // branch-free speculative prefetch of the tile after next
    {
      const int kpf = (k0 + 64 < CC) ? (k0 + 64) : 0;
      const int r = tid >> 1;
      __builtin_prefetch(&A [(size_t)(m0 + r) * CC + kpf], 0, 0);
      __builtin_prefetch(&Bt[(size_t)(n0 + r) * CC + kpf], 0, 0);
    }
    // compute from buffer p
    bf16x16 a0 = load_frag16(&As[p][wr * 32][0],      40, lane);
    bf16x16 a1 = load_frag16(&As[p][wr * 32 + 16][0], 40, lane);
#pragma unroll
    for (int nt = 0; nt < 4; ++nt) {
      bf16x16 b = load_frag16(&Bs[p][wc * 64 + nt * 16][0], 40, lane);
      acc[nt]     = wmma_bf16(a0, b, acc[nt]);
      acc[4 + nt] = wmma_bf16(a1, b, acc[4 + nt]);
    }
    // stage next tile into the other buffer; single barrier per K-step
    lstore(p ^ 1);
    __syncthreads();
    p ^= 1;
  }

  const int nL = lane & 15;
  const int mh = (lane >> 4) << 3;
#pragma unroll
  for (int mt = 0; mt < 2; ++mt) {
#pragma unroll
    for (int nt = 0; nt < 4; ++nt) {
      const int n = n0 + wc * 64 + nt * 16 + nL;
      const float bn = bias[n];
      const f32x8 a = acc[mt * 4 + nt];
#pragma unroll
      for (int v = 0; v < 8; ++v) {
        const int m = m0 + wr * 32 + mt * 16 + mh + v;
        if (MODE == 0) {
          Cout[(size_t)m * Nld + n] = a[v] + bn;
        } else {
          const int t = m >> 9;   // m / 512
          const int s = m & 511;  // m % 512
          const size_t oi = ((size_t)t * CC + n) * SS + s;
          Cout[oi] = a[v] + bn + resid[oi];
        }
      }
    }
  }
}

// ---------------------------------------------------------------------------
// Attention: one workgroup per (s, he).  T=64, HD=64.
// LN(Q)/LN(K) (1/sqrt(HD) folded into Q), S = Q K^T + bias, softmax, O = P V.
// ---------------------------------------------------------------------------
__global__ __launch_bounds__(128)
void attn_kernel(const float* __restrict__ qkv,
                 const float* __restrict__ qw, const float* __restrict__ qb,
                 const float* __restrict__ kw, const float* __restrict__ kbv,
                 const float* __restrict__ biasT, __bf16* __restrict__ obuf) {
  const int s  = blockIdx.x;  // 0..511
  const int he = blockIdx.y;  // 0..11
  __shared__ __bf16 Qs[64][72];  // [t][hd]
  __shared__ __bf16 Ks[64][72];  // [t'][hd]   (NT B operand for S)
  __shared__ __bf16 Vt[64][72];  // [hd][t']   (NT B operand for O)
  __shared__ float  Sm[64][65];
  __shared__ __bf16 Ps[64][72];  // softmax probs, A operand for O
  const int tid  = threadIdx.x;
  const int lane = tid & 31;
  const int wv   = tid >> 5;

  // Phase 1: load + layernorm.  tid<64 -> Q row t=tid; tid>=64 -> K row t=tid-64.
  {
    const int t = tid & 63;
    const size_t rowbase = ((size_t)(t * SS + s)) * N_QKV + (size_t)he * 192;
    const float* src = qkv + rowbase + ((tid < 64) ? 0 : HD);
    float vbuf[HD];
    float mu = 0.f;
#pragma unroll 8
    for (int i = 0; i < HD; ++i) { vbuf[i] = src[i]; mu += vbuf[i]; }
    mu *= (1.f / HD);
    float var = 0.f;
#pragma unroll 8
    for (int i = 0; i < HD; ++i) { float d = vbuf[i] - mu; var += d * d; }
    const float inv = rsqrtf(var * (1.f / HD) + 1e-5f);
    if (tid < 64) {
#pragma unroll 8
      for (int i = 0; i < HD; ++i)
        Qs[t][i] = (__bf16)((((vbuf[i] - mu) * inv) * qw[i] + qb[i]) * 0.125f);
    } else {
#pragma unroll 8
      for (int i = 0; i < HD; ++i)
        Ks[t][i] = (__bf16)(((vbuf[i] - mu) * inv) * kw[i] + kbv[i]);
    }
    // V: thread t loads hd 0..31, thread t+64 loads hd 32..63 (transposed store)
    const float* vsrc = qkv + rowbase + 2 * HD;
    const int h0 = (tid < 64) ? 0 : 32;
#pragma unroll 8
    for (int i = 0; i < 32; ++i) Vt[h0 + i][t] = (__bf16)vsrc[h0 + i];
  }
  __syncthreads();

  // Phase 2: S = Q K^T + bias  (wave wv -> 16-row strip)
  {
    f32x8 acc[4] = {};
#pragma unroll
    for (int kk = 0; kk < HD; kk += 32) {
      bf16x16 aq = load_frag16(&Qs[wv * 16][kk], 72, lane);
#pragma unroll
      for (int nt = 0; nt < 4; ++nt)
        acc[nt] = wmma_bf16(aq, load_frag16(&Ks[nt * 16][kk], 72, lane), acc[nt]);
    }
    const float* bptr = biasT + (size_t)he * (TT * TT);
    const int nL = lane & 15;
    const int mh = (lane >> 4) << 3;
#pragma unroll
    for (int nt = 0; nt < 4; ++nt) {
      const int n = nt * 16 + nL;
#pragma unroll
      for (int v = 0; v < 8; ++v) {
        const int m = wv * 16 + mh + v;
        Sm[m][n] = acc[nt][v] + bptr[m * TT + n];
      }
    }
  }
  __syncthreads();

  // Phase 3: softmax over rows (threads 0..63)
  if (tid < 64) {
    float mx = -1e30f;
#pragma unroll 8
    for (int j = 0; j < TT; ++j) mx = fmaxf(mx, Sm[tid][j]);
    float sum = 0.f;
#pragma unroll 8
    for (int j = 0; j < TT; ++j) {
      float e = __expf(Sm[tid][j] - mx);
      Sm[tid][j] = e;
      sum += e;
    }
    const float r = 1.f / sum;
#pragma unroll 8
    for (int j = 0; j < TT; ++j) Ps[tid][j] = (__bf16)(Sm[tid][j] * r);
  }
  __syncthreads();

  // Phase 4: O = P V, write bf16 obuf[m = t*512+s][c = he*64+hd]
  {
    f32x8 acc[4] = {};
#pragma unroll
    for (int kk = 0; kk < TT; kk += 32) {
      bf16x16 ap = load_frag16(&Ps[wv * 16][kk], 72, lane);
#pragma unroll
      for (int nt = 0; nt < 4; ++nt)
        acc[nt] = wmma_bf16(ap, load_frag16(&Vt[nt * 16][kk], 72, lane), acc[nt]);
    }
    const int nL = lane & 15;
    const int mh = (lane >> 4) << 3;
#pragma unroll
    for (int nt = 0; nt < 4; ++nt) {
      const int hd = nt * 16 + nL;
#pragma unroll
      for (int v = 0; v < 8; ++v) {
        const int t = wv * 16 + mh + v;
        obuf[((size_t)(t * SS + s)) * CC + he * HD + hd] = (__bf16)acc[nt][v];
      }
    }
  }
}

// ---------------------------------------------------------------------------
// Launcher
// ---------------------------------------------------------------------------
extern "C" void kernel_launch(void* const* d_in, const int* in_sizes, int n_in,
                              void* d_out, int out_size, void* d_ws, size_t ws_size,
                              hipStream_t stream) {
  const float* x    = (const float*)d_in[0];   // (T,B,C,H,W,D)
  const float* nw   = (const float*)d_in[1];   // (C,)
  const float* in_w = (const float*)d_in[2];   // (3C, C)
  const float* in_b = (const float*)d_in[3];   // (3C,)
  const float* ow   = (const float*)d_in[4];   // (C, C)
  const float* ob   = (const float*)d_in[5];   // (C,)
  const float* qlw  = (const float*)d_in[6];
  const float* qlb  = (const float*)d_in[7];
  const float* klw  = (const float*)d_in[8];
  const float* klb  = (const float*)d_in[9];
  const float* rel  = (const float*)d_in[10];  // (32, HE)
  float* out = (float*)d_out;

  // Workspace layout (all offsets 256B aligned)
  char* ws = (char*)d_ws;
  __bf16* xnA    = (__bf16*)(ws);                                   // 32768*768*2  = 50331648
  __bf16* wqkvB  = (__bf16*)(ws + 50331648);                        // 2304*768*2   = 3538944
  __bf16* woutB  = (__bf16*)(ws + 50331648 + 3538944);              // 768*768*2    = 1179648
  float*  biasT  = (float*) (ws + 50331648 + 3538944 + 1179648);    // 12*64*64*4   = 196608
  float*  qkvbuf = (float*) (ws + 55246848);                        // 32768*2304*4 = 301989888
  __bf16* obuf   = (__bf16*)(ws + 55246848 + 301989888);            // 32768*768*2

  // 1) weight casts to bf16 (layouts already (N,K) row-major -> NT GEMM)
  {
    int n1 = N_QKV * CC;
    cast_bf16_kernel<<<(n1 + 255) / 256, 256, 0, stream>>>(in_w, wqkvB, n1);
    int n2 = CC * CC;
    cast_bf16_kernel<<<(n2 + 255) / 256, 256, 0, stream>>>(ow, woutB, n2);
  }
  // 2) T5 relative position bias table
  relbias_kernel<<<TT, TT, 0, stream>>>(rel, biasT);
  // 3) RMS norm + transpose to activation matrix
  rmsnorm_kernel<<<dim3(CC, TT), 128, 0, stream>>>(x, nw, xnA);
  // 4) QKV projection GEMM (32768 x 2304 x 768), 128x128 tiles
  gemm_nt_kernel<0><<<dim3(MM / 128, N_QKV / 128), 256, 0, stream>>>(
      xnA, wqkvB, in_b, nullptr, qkvbuf, N_QKV);
  // 5) attention per (spatial, head)
  attn_kernel<<<dim3(SS, HE), 128, 0, stream>>>(qkvbuf, qlw, qlb, klw, klb, biasT, obuf);
  // 6) output projection GEMM + bias + residual, scatter to (T,C,S)
  gemm_nt_kernel<1><<<dim3(MM / 128, CC / 128), 256, 0, stream>>>(
      obuf, woutB, ob, x, out, CC);
}
